// Attention_57449482551335
// MI455X (gfx1250) — matrix operand order repaired and checked by reading, not compiled, not run on hardware
//
#include <hip/hip_runtime.h>
#include <hip/hip_bf16.h>

// ---------------------------------------------------------------------------
// FAVOR+ causal linear attention, chunked, WMMA f16->f32 on gfx1250 (wave32).
// B=2, L=4096, H=8, D=64, M=128. Chunk C=64, NC=64, augmented D (ones col)=80.
// All matmuls on v_wmma_f32_16x16x32_f16; V staged to LDS transposed f16 so
// every B-fragment is a single 32B ds load.
// ---------------------------------------------------------------------------

typedef _Float16 h8   __attribute__((ext_vector_type(8)));
typedef _Float16 v16h __attribute__((ext_vector_type(16)));
typedef float    v8f  __attribute__((ext_vector_type(8)));

constexpr int Bb   = 2;
constexpr int Ll   = 4096;
constexpr int Hh   = 8;
constexpr int Dd   = 64;
constexpr int Mm   = 128;
constexpr int Cc   = 64;            // chunk length
constexpr int NC   = Ll / Cc;       // 64 chunks
constexpr int BH   = Bb * Hh;       // 16
constexpr int DAUG = 80;            // D + 16 (col 64 = ones -> k-sum / denominator)
constexpr int HD   = Hh * Dd;       // token stride in [B,L,H,D] = 512
constexpr float RATIO = 0.08838834764831845f;   // 1/sqrt(M)
constexpr float EPS   = 1e-3f;

static __device__ __forceinline__ v16h mk16(h8 lo, h8 hi) {
  return __builtin_shufflevector(lo, hi, 0,1,2,3,4,5,6,7,8,9,10,11,12,13,14,15);
}

// A fragment 16x32 f16 from row-major f32 (row stride ldr): lane=row, kb per half-wave.
static __device__ __forceinline__ v16h a_frag_f32(const float* base, int ldr, int lane, int kbase) {
  const float* p = base + (size_t)(lane & 15) * ldr + kbase + ((lane >> 4) << 4);
  h8 lo, hi;
#pragma unroll
  for (int i = 0; i < 8; ++i) { lo[i] = (_Float16)p[i]; hi[i] = (_Float16)p[8 + i]; }
  return mk16(lo, hi);
}

// A fragment 16x32 f16 from row-major f16 (LDS).
static __device__ __forceinline__ v16h a_frag_f16(const _Float16* base, int ldr, int lane, int kbase) {
  const _Float16* p = base + (size_t)(lane & 15) * ldr + kbase + ((lane >> 4) << 4);
  h8 lo = *(const h8*)p;
  h8 hi = *(const h8*)(p + 8);
  return mk16(lo, hi);
}

// B fragment 32x16 f16 from column-major f32 storage [col][k] (col stride ldc): lane=col.
static __device__ __forceinline__ v16h b_frag_f32(const float* base, int ldc, int lane, int kbase) {
  const float* p = base + (size_t)(lane & 15) * ldc + kbase + ((lane >> 4) << 4);
  v16h r;
#pragma unroll
  for (int i = 0; i < 16; ++i) r[i] = (_Float16)p[i];
  return r;
}

// B fragment 32x16 f16 from column-major f16 storage [col][k]: one 32B load per lane.
static __device__ __forceinline__ v16h b_frag_f16(const _Float16* base, int ldc, int lane, int kbase) {
  const _Float16* p = base + (size_t)(lane & 15) * ldc + kbase + ((lane >> 4) << 4);
  return *(const v16h*)p;
}

// Synthetic augmented tile: column 0 (global col 64) = ones, rest zeros.
static __device__ __forceinline__ v16h b_frag_ones(int lane) {
  _Float16 v = ((lane & 15) == 0) ? (_Float16)1.0f : (_Float16)0.0f;
  v16h r;
#pragma unroll
  for (int i = 0; i < 16; ++i) r[i] = v;
  return r;
}

static __device__ __forceinline__ v8f wmma16(v16h a, v16h b, v8f c) {
  return __builtin_amdgcn_wmma_f32_16x16x32_f16(false, a, false, b, (short)0, c, false, false);
}

// Cooperative stage: V chunk [C][D] f32 (token stride HD) -> LDS f16 transposed [d][t].
static __device__ __forceinline__ void stage_vt(const float* vb, _Float16* sVt, int tid) {
#pragma unroll
  for (int j = 0; j < (Cc * Dd) / 256; ++j) {
    const int e = tid + j * 256;
    const int t = e >> 6, d = e & 63;            // consecutive tid -> consecutive d: coalesced
    sVt[d * Cc + t] = (_Float16)vb[(size_t)t * HD + d];
  }
}

// ---------------------------------------------------------------------------
// Kernel 1: per-(bh,chunk)  KVc_aug[d][m] = sum_{t in chunk} Kp[t][m] * Vaug[t][d]
// ---------------------------------------------------------------------------
__global__ __launch_bounds__(256) void favor_chunk_kv(const float* __restrict__ K,
                                                      const float* __restrict__ V,
                                                      const float* __restrict__ P,
                                                      _Float16* __restrict__ kvc) {
  __shared__ __align__(32) _Float16 sKpT[Mm * Cc];   // [m][t] 16 KB
  __shared__ __align__(32) _Float16 sVt [Dd * Cc];   // [d][t]  8 KB
  const int chunk = blockIdx.x, bh = blockIdx.y;
  const int b = bh >> 3, h = bh & 7;
  const int tid = threadIdx.x, lane = tid & 31, wave = tid >> 5;
  const int cl = lane & 15, mbase = (lane >> 4) << 3;
  const float* kb_p = K + ((size_t)(b * Ll + chunk * Cc) * Hh + h) * Dd;
  const float* vb_p = V + ((size_t)(b * Ll + chunk * Cc) * Hh + h) * Dd;

  __builtin_prefetch(kb_p, 0, 3);            // global_prefetch_b8
  __builtin_prefetch(vb_p, 0, 3);

  // Stage 0: V chunk -> LDS [d][t] f16
  stage_vt(vb_p, sVt, tid);

  // Stage 1: Kp = relu(ratio * K_chunk @ P^T) + eps  -> LDS transposed [m][t]
  for (int tile = wave; tile < 32; tile += 8) {
    const int tt = tile >> 3, mt = tile & 7;
    v8f acc = {0.f,0.f,0.f,0.f,0.f,0.f,0.f,0.f};
#pragma unroll
    for (int kb = 0; kb < Dd; kb += 32) {
      v16h a  = a_frag_f32(kb_p + (size_t)tt * 16 * HD, HD, lane, kb);
      v16h bb = b_frag_f32(P + (size_t)mt * 16 * Dd, Dd, lane, kb);
      acc = wmma16(a, bb, acc);
    }
    h8 pk;
#pragma unroll
    for (int r = 0; r < 8; ++r) {
      float x = RATIO * acc[r];
      pk[r] = (_Float16)((x > 0.f ? x : 0.f) + EPS);
    }
    // C-frag: row t = tt*16+mbase+r (contiguous in r), col m = mt*16+cl
    *(h8*)&sKpT[(size_t)(mt * 16 + cl) * Cc + tt * 16 + mbase] = pk;
  }
  __syncthreads();

  // Stage 2: KVc tiles (8 m-tiles x 5 d-tiles), stored [d][m] f16
  for (int tile = wave; tile < 40; tile += 8) {
    const int mt = tile / 5, dt = tile % 5;
    v8f acc = {0.f,0.f,0.f,0.f,0.f,0.f,0.f,0.f};
#pragma unroll
    for (int kt = 0; kt < Cc; kt += 32) {
      v16h a  = a_frag_f16(sKpT + (size_t)mt * 16 * Cc, Cc, lane, kt);
      v16h bb = (dt < 4) ? b_frag_f16(sVt + (size_t)(dt * 16) * Cc, Cc, lane, kt)
                         : b_frag_ones(lane);
      acc = wmma16(a, bb, acc);
    }
    h8 pk;
#pragma unroll
    for (int r = 0; r < 8; ++r) pk[r] = (_Float16)acc[r];
    // C-frag row m = mt*16+mbase+r contiguous, col d = dt*16+cl
    _Float16* dst = kvc + ((size_t)(bh * NC + chunk) * DAUG + dt * 16 + cl) * Mm + mt * 16 + mbase;
    *(h8*)dst = pk;
  }
}

// ---------------------------------------------------------------------------
// Kernel 2: exclusive prefix sum over chunks of the KV_aug states (f32 accum)
// ---------------------------------------------------------------------------
__global__ __launch_bounds__(256) void favor_scan(const _Float16* __restrict__ kvc,
                                                  _Float16* __restrict__ kvs) {
  const int bh = blockIdx.y;
  const int e = blockIdx.x * 256 + threadIdx.x;   // 0..10239 element of [DAUG][M]
  float st = 0.f;
  for (int c = 0; c < NC; ++c) {
    const size_t off = (size_t)(bh * NC + c) * (DAUG * Mm) + e;
    kvs[off] = (_Float16)st;
    st += (float)kvc[off];
  }
}

// ---------------------------------------------------------------------------
// Kernel 3: out_aug = Qp @ KVstate + tril(Qp Kp^T) @ Vaug ; divide by col 64
// ---------------------------------------------------------------------------
__global__ __launch_bounds__(256) void favor_out_kernel(const float* __restrict__ Q,
                                                        const float* __restrict__ K,
                                                        const float* __restrict__ V,
                                                        const float* __restrict__ P,
                                                        const _Float16* __restrict__ kvs,
                                                        float* __restrict__ out) {
  __shared__ __align__(32) _Float16 sQp[Cc * Mm];   // [t][m] 16 KB
  __shared__ __align__(32) float    sND[Cc * DAUG]; // 20 KB; aliased as sKp (16 KB) pre-barrier
  __shared__ __align__(32) _Float16 sS [Cc * Cc];   // [t][t'] 8 KB
  __shared__ __align__(32) _Float16 sVt[Dd * Cc];   // [d][t]  8 KB
  _Float16* sKp = (_Float16*)sND;                    // dead after stage 2 -> reuse for sND

  const int chunk = blockIdx.x, bh = blockIdx.y;
  const int b = bh >> 3, h = bh & 7;
  const int tid = threadIdx.x, lane = tid & 31, wave = tid >> 5;
  const int cl = lane & 15, mbase = (lane >> 4) << 3;
  const float* qb   = Q + ((size_t)(b * Ll + chunk * Cc) * Hh + h) * Dd;
  const float* kb_p = K + ((size_t)(b * Ll + chunk * Cc) * Hh + h) * Dd;
  const float* vb   = V + ((size_t)(b * Ll + chunk * Cc) * Hh + h) * Dd;

  __builtin_prefetch(qb, 0, 3);              // global_prefetch_b8
  __builtin_prefetch(kb_p, 0, 3);
  __builtin_prefetch(vb, 0, 3);

  // zero S (upper-triangle tiles are never written)
  for (int e2 = tid; e2 < (Cc * Cc) / 2; e2 += 256) ((unsigned int*)sS)[e2] = 0u;

  // Stage 0: V chunk -> LDS [d][t] f16
  stage_vt(vb, sVt, tid);

  // Stage 1: Qp and Kp features -> LDS row-major [t][m]
  for (int tile = wave; tile < 64; tile += 8) {
    const bool isQ = tile < 32;
    const int t2 = tile & 31, tt = t2 >> 3, mt = t2 & 7;
    const float* src = isQ ? qb : kb_p;
    _Float16* dst = isQ ? sQp : sKp;
    v8f acc = {0.f,0.f,0.f,0.f,0.f,0.f,0.f,0.f};
#pragma unroll
    for (int kb = 0; kb < Dd; kb += 32) {
      v16h a  = a_frag_f32(src + (size_t)tt * 16 * HD, HD, lane, kb);
      v16h bb = b_frag_f32(P + (size_t)mt * 16 * Dd, Dd, lane, kb);
      acc = wmma16(a, bb, acc);
    }
#pragma unroll
    for (int r = 0; r < 8; ++r) {
      float x = RATIO * acc[r];
      dst[(size_t)(tt * 16 + mbase + r) * Mm + mt * 16 + cl] =
          (_Float16)((x > 0.f ? x : 0.f) + EPS);
    }
  }
  __syncthreads();

  // Stage 2: S = tril(Qp @ Kp^T) over 4x4 tiles (lower triangle only)
  for (int i = wave; i < 16; i += 8) {
    const int it = i >> 2, jt = i & 3;
    if (jt <= it) {
      v8f acc = {0.f,0.f,0.f,0.f,0.f,0.f,0.f,0.f};
#pragma unroll
      for (int km = 0; km < Mm; km += 32) {
        v16h a  = a_frag_f16(sQp + (size_t)it * 16 * Mm, Mm, lane, km);
        v16h bb = b_frag_f16(sKp + (size_t)jt * 16 * Mm, Mm, lane, km);
        acc = wmma16(a, bb, acc);
      }
#pragma unroll
      for (int r = 0; r < 8; ++r) {
        float v = acc[r];
        if (it == jt && cl > mbase + r) v = 0.f;   // causal mask on diagonal tile
        sS[(size_t)(it * 16 + mbase + r) * Cc + jt * 16 + cl] = (_Float16)v;
      }
    }
  }
  __syncthreads();   // after this barrier sKp is dead; its storage becomes sND

  // Stage 3: out_aug tiles = Qp @ KVstate  +  S @ Vaug
  const _Float16* kvsb = kvs + (size_t)(bh * NC + chunk) * (DAUG * Mm);
  for (int tile = wave; tile < 20; tile += 8) {
    const int tt = tile / 5, dt = tile % 5;
    v8f acc = {0.f,0.f,0.f,0.f,0.f,0.f,0.f,0.f};
#pragma unroll
    for (int km = 0; km < Mm; km += 32) {            // inter-chunk
      v16h a  = a_frag_f16(sQp + (size_t)tt * 16 * Mm, Mm, lane, km);
      v16h bb = b_frag_f16(kvsb + (size_t)(dt * 16) * Mm, Mm, lane, km);
      acc = wmma16(a, bb, acc);
    }
#pragma unroll
    for (int kt = 0; kt < Cc; kt += 32) {            // intra-chunk
      v16h a  = a_frag_f16(sS + (size_t)tt * 16 * Cc, Cc, lane, kt);
      v16h bb = (dt < 4) ? b_frag_f16(sVt + (size_t)(dt * 16) * Cc, Cc, lane, kt)
                         : b_frag_ones(lane);
      acc = wmma16(a, bb, acc);
    }
#pragma unroll
    for (int r = 0; r < 8; ++r)
      sND[(size_t)(tt * 16 + mbase + r) * DAUG + dt * 16 + cl] = acc[r];
  }
  __syncthreads();

  // Stage 4: out[t][d] = num / den   (den = augmented column 64)
  for (int e2 = tid; e2 < Cc * Dd; e2 += 256) {
    const int t = e2 >> 6, d = e2 & 63;
    const float den = sND[(size_t)t * DAUG + 64];
    out[((size_t)(b * Ll + chunk * Cc + t) * Hh + h) * Dd + d] =
        sND[(size_t)t * DAUG + d] / den;
  }
}

// ---------------------------------------------------------------------------
extern "C" void kernel_launch(void* const* d_in, const int* in_sizes, int n_in,
                              void* d_out, int out_size, void* d_ws, size_t ws_size,
                              hipStream_t stream) {
  (void)in_sizes; (void)n_in; (void)out_size; (void)ws_size;
  const float* Q = (const float*)d_in[0];
  const float* K = (const float*)d_in[1];
  const float* V = (const float*)d_in[2];
  const float* P = (const float*)d_in[3];
  float* out = (float*)d_out;

  _Float16* kvc = (_Float16*)d_ws;                                // per-chunk KV sums
  _Float16* kvs = kvc + (size_t)BH * NC * DAUG * Mm;              // exclusive prefix states

  favor_chunk_kv<<<dim3(NC, BH), 256, 0, stream>>>(K, V, P, kvc);
  favor_scan<<<dim3((DAUG * Mm) / 256, BH), 256, 0, stream>>>(kvc, kvs);
  favor_out_kernel<<<dim3(NC, BH), 256, 0, stream>>>(Q, K, V, P, kvs, out);
}
